// GraphCorrelationLayer_65403761983862
// MI455X (gfx1250) — compile-verified, matrix operand order checked by reading, not run on hardware
//
#include <hip/hip_runtime.h>
#include <hip/hip_bf16.h>

typedef __attribute__((ext_vector_type(16))) _Float16 v16h;
typedef __attribute__((ext_vector_type(8)))  _Float16 v8h;
typedef __attribute__((ext_vector_type(4)))  _Float16 v4h;
typedef __attribute__((ext_vector_type(8)))  float    v8f;

// ---- LDS layout (bytes), total 148992 (< 320KB/WGP, 2 blocks/WGP resident) ----
#define OFF_XN   0            // f16 [64][136]  normalized x
#define OFF_XH   17408        // f16 [64][136]  raw x as f16
#define OFF_W    34816        // f16 [128][136] weight matrix
#define OFF_CORR 69632        // f32 [64][68]   correlation
#define OFF_WTS  87040        // f16 [64][72]   softmax weights
#define OFF_HT   96256        // f16 [128][72]  h transposed: ht[o][c]
#define OFF_OL   114688       // f32 [64][132]  pre-LN output
#define OFF_B    148480       // f32 [128]      bias
#define SMEM_BYTES 148992

__device__ __forceinline__ v8f wmma16(v16h a, v16h b, v8f c) {
    return __builtin_amdgcn_wmma_f32_16x16x32_f16(
        /*neg_a=*/false, a, /*neg_b=*/false, b,
        /*c_mod=*/(short)0, c, /*reuse_a=*/false, /*reuse_b=*/false);
}

// Load one 16x32 f16 fragment (A layout; identical gather works for B when B is
// the transpose of a row-major LDS matrix). stride in halves, 16B-aligned rows.
__device__ __forceinline__ v16h ld_frag(const _Float16* __restrict__ p0,
                                        int row_base, int stride, int kbase) {
    const int lane = threadIdx.x & 31;
    const int r  = row_base + (lane & 15);
    const int kh = (lane >> 4) << 3;              // 0 or 8
    const _Float16* p = p0 + r * stride + kbase + kh;
    v8h lo = *(const v8h*)(p);
    v8h hi = *(const v8h*)(p + 16);
    v16h f;
#pragma unroll
    for (int i = 0; i < 8; ++i) { f[i] = lo[i]; f[i + 8] = hi[i]; }
    return f;
}

__global__ __launch_bounds__(256) void gcl_kernel(
    const float* __restrict__ x, const float* __restrict__ W,
    const float* __restrict__ bvec, const float* __restrict__ gamma,
    const float* __restrict__ beta, float* __restrict__ out) {

    extern __shared__ char smem[];
    _Float16* xn   = (_Float16*)(smem + OFF_XN);
    _Float16* xh   = (_Float16*)(smem + OFF_XH);
    _Float16* wl   = (_Float16*)(smem + OFF_W);
    float*    corr = (float*)   (smem + OFF_CORR);
    _Float16* wts  = (_Float16*)(smem + OFF_WTS);
    _Float16* ht   = (_Float16*)(smem + OFF_HT);
    float*    ol   = (float*)   (smem + OFF_OL);
    float*    bl   = (float*)   (smem + OFF_B);

    const int  t    = threadIdx.x;
    const int  wave = t >> 5;
    const int  lane = t & 31;
    const long bt   = blockIdx.x;
    const float* xg = x + bt * (64 * 128);

    // ---- Phase 1a: W -> LDS f16, bias -> LDS ----
    {
        const float4* W4 = (const float4*)W;
        for (int g = t; g < 4096; g += 256) {      // 128*128/4 float4s
            float4 w4 = W4[g];
            int o = g >> 5;                        // 32 float4 per row
            int f = (g & 31) << 2;
            v4h h4 = { (_Float16)w4.x, (_Float16)w4.y, (_Float16)w4.z, (_Float16)w4.w };
            *(v4h*)(wl + o * 136 + f) = h4;
        }
        if (t < 128) bl[t] = bvec[t];
    }

    // ---- Phase 1b: load x, per-channel stats, write Xn(f16) + X(f16) ----
    {
        const int row = t >> 2, q = t & 3;         // 4 threads per channel row
        const float4* xr = (const float4*)(xg + row * 128 + q * 32);
        float4 v[8];
        float s1 = 0.f, s2 = 0.f;
#pragma unroll
        for (int i = 0; i < 8; ++i) {
            v[i] = xr[i];
            s1 += v[i].x + v[i].y + v[i].z + v[i].w;
            s2 += v[i].x * v[i].x + v[i].y * v[i].y + v[i].z * v[i].z + v[i].w * v[i].w;
        }
        s1 += __shfl_xor(s1, 1, 32); s2 += __shfl_xor(s2, 1, 32);
        s1 += __shfl_xor(s1, 2, 32); s2 += __shfl_xor(s2, 2, 32);
        float mean = s1 * (1.f / 128.f);
        float var  = (s2 - 128.f * mean * mean) * (1.f / 127.f);   // unbiased
        var = fmaxf(var, 0.f);
        float inv = 1.f / (sqrtf(var) + 1e-6f);
#pragma unroll
        for (int i = 0; i < 8; ++i) {
            float xs[4] = { v[i].x, v[i].y, v[i].z, v[i].w };
            v4h hn, hh;
#pragma unroll
            for (int j = 0; j < 4; ++j) {
                hn[j] = (_Float16)((xs[j] - mean) * inv);
                hh[j] = (_Float16)xs[j];
            }
            int c = q * 32 + i * 4;
            *(v4h*)(xn + row * 136 + c) = hn;
            *(v4h*)(xh + row * 136 + c) = hh;
        }
    }
    __syncthreads();

    // ---- Phase 2a: corr = Xn * Xn^T / 128 (16 tiles, 2/wave) ----
#pragma unroll
    for (int s = 0; s < 2; ++s) {
        int idx = wave * 2 + s;
        int ti = idx >> 2, tj = idx & 3;
        v8f acc = {};
#pragma unroll
        for (int kc = 0; kc < 4; ++kc) {
            v16h a  = ld_frag(xn, ti * 16, 136, kc * 32);
            v16h bf = ld_frag(xn, tj * 16, 136, kc * 32);
            acc = wmma16(a, bf, acc);
        }
        int n  = tj * 16 + (lane & 15);
        int mb = ti * 16 + ((lane >> 4) << 3);
#pragma unroll
        for (int r = 0; r < 8; ++r)
            corr[(mb + r) * 68 + n] = acc[r] * (1.f / 128.f);
    }

    // ---- Phase 2b: h = X*W^T + b (32 tiles, 4/wave), store transposed ----
#pragma unroll
    for (int s = 0; s < 4; ++s) {
        int idx = wave * 4 + s;
        int ti = idx >> 3, to = idx & 7;
        v8f acc = {};
#pragma unroll
        for (int kc = 0; kc < 4; ++kc) {
            v16h a  = ld_frag(xh, ti * 16, 136, kc * 32);
            v16h bf = ld_frag(wl, to * 16, 136, kc * 32);   // W row o, contiguous in f
            acc = wmma16(a, bf, acc);
        }
        int o    = to * 16 + (lane & 15);
        float bb = bl[o];
        int mb   = ti * 16 + ((lane >> 4) << 3);
#pragma unroll
        for (int r = 0; r < 8; ++r)
            ht[o * 72 + mb + r] = (_Float16)(acc[r] + bb);
    }
    __syncthreads();

    // ---- Phase 3: softmax over corr rows -> wts f16 ----
    {
        const int row = t >> 2, q = t & 3;         // 4 threads x 16 cols each
        float vals[16];
        float mx = -1e30f;
#pragma unroll
        for (int i = 0; i < 16; ++i) {
            vals[i] = corr[row * 68 + q * 16 + i];
            mx = fmaxf(mx, vals[i]);
        }
        mx = fmaxf(mx, __shfl_xor(mx, 1, 32));
        mx = fmaxf(mx, __shfl_xor(mx, 2, 32));
        float sum = 0.f;
#pragma unroll
        for (int i = 0; i < 16; ++i) { vals[i] = __expf(vals[i] - mx); sum += vals[i]; }
        sum += __shfl_xor(sum, 1, 32);
        sum += __shfl_xor(sum, 2, 32);
        float rs = 1.f / sum;
#pragma unroll
        for (int i = 0; i < 16; i += 4) {
            v4h h4;
#pragma unroll
            for (int j = 0; j < 4; ++j) h4[j] = (_Float16)(vals[i + j] * rs);
            *(v4h*)(wts + row * 72 + q * 16 + i) = h4;
        }
    }
    __syncthreads();

    // ---- Phase 4: out = weights * h (32 tiles, 4/wave, K=64) ----
#pragma unroll
    for (int s = 0; s < 4; ++s) {
        int idx = wave * 4 + s;
        int ti = idx >> 3, to = idx & 7;
        v8f acc = {};
#pragma unroll
        for (int kc = 0; kc < 2; ++kc) {
            v16h a  = ld_frag(wts, ti * 16, 72, kc * 32);
            v16h bf = ld_frag(ht,  to * 16, 72, kc * 32);   // ht[o][j] contiguous in j
            acc = wmma16(a, bf, acc);
        }
        int n  = to * 16 + (lane & 15);
        int mb = ti * 16 + ((lane >> 4) << 3);
#pragma unroll
        for (int r = 0; r < 8; ++r)
            ol[(mb + r) * 132 + n] = acc[r];
    }
    __syncthreads();

    // ---- Phase 5: LayerNorm (biased var, eps=1e-5) + ReLU -> global ----
    {
        const int row = t >> 2, q = t & 3;
        float vals[32];
        float s1 = 0.f, s2 = 0.f;
#pragma unroll
        for (int i = 0; i < 32; ++i) {
            float v = ol[row * 132 + q * 32 + i];
            vals[i] = v; s1 += v; s2 += v * v;
        }
        s1 += __shfl_xor(s1, 1, 32); s2 += __shfl_xor(s2, 1, 32);
        s1 += __shfl_xor(s1, 2, 32); s2 += __shfl_xor(s2, 2, 32);
        float mu  = s1 * (1.f / 128.f);
        float var = s2 * (1.f / 128.f) - mu * mu;
        float rs  = rsqrtf(var + 1e-5f);
        float* og = out + bt * (64 * 128) + row * 128 + q * 32;
#pragma unroll
        for (int i = 0; i < 32; i += 4) {
            float4 o4;
            float* ov = &o4.x;
#pragma unroll
            for (int j = 0; j < 4; ++j) {
                int c = q * 32 + i + j;
                float val = (vals[i + j] - mu) * rs * gamma[c] + beta[c];
                ov[j] = fmaxf(val, 0.f);
            }
            *(float4*)(og + i) = o4;
        }
    }
}

extern "C" void kernel_launch(void* const* d_in, const int* in_sizes, int n_in,
                              void* d_out, int out_size, void* d_ws, size_t ws_size,
                              hipStream_t stream) {
    (void)n_in; (void)out_size; (void)d_ws; (void)ws_size;
    const float* x     = (const float*)d_in[0];
    const float* W     = (const float*)d_in[1];
    const float* bvec  = (const float*)d_in[2];
    const float* gamma = (const float*)d_in[3];
    const float* beta  = (const float*)d_in[4];
    float* out = (float*)d_out;

    const int bt = in_sizes[0] / (64 * 128);       // B*T = 6400
    hipLaunchKernelGGL(gcl_kernel, dim3(bt), dim3(256), SMEM_BYTES, stream,
                       x, W, bvec, gamma, beta, out);
}
